// ConvBertSelfAttention_47983374631651
// MI455X (gfx1250) — matrix operand order repaired and checked by reading, not compiled, not run on hardware
//
#include <hip/hip_runtime.h>
#include <hip/hip_bf16.h>

typedef _Float16 f16;
typedef __attribute__((ext_vector_type(8)))  _Float16     v8h;
typedef __attribute__((ext_vector_type(16))) _Float16     v16h;
typedef __attribute__((ext_vector_type(8)))  float        v8f;
typedef __attribute__((ext_vector_type(4)))  unsigned int v4u;
typedef __attribute__((ext_vector_type(4)))  int          v4i;
typedef __attribute__((ext_vector_type(8)))  int          v8i;

#define B_   4
#define S_   2048
#define H_   8
#define D_   64
#define A_   512
#define HID_ 1024
#define KW_  9
#define M_   (B_*S_)
#define NCK_ (H_*KW_)   // 72

// Build a 16-element f16 fragment from two contiguous 8-element (16B) runs.
__device__ __forceinline__ v16h ldfrag(const f16* p0, const f16* p1) {
  union { v16h v; v8h h[2]; } u;
  u.h[0] = *(const v8h*)p0;
  u.h[1] = *(const v8h*)p1;
  return u.v;
}

__device__ __forceinline__ v8f wmma_f16(v16h a, v16h b, v8f c) {
  return __builtin_amdgcn_wmma_f32_16x16x32_f16(false, a, false, b, (short)0, c,
                                                false, false);
}

// ---------------- weight conversion ----------------
__global__ void cvt_f16(const float* __restrict__ src, f16* __restrict__ dst, int n) {
  int i = blockIdx.x * 256 + threadIdx.x;
  if (i < n) dst[i] = (f16)src[i];
}

// Wck (72 x 512) -> zero-padded (128 x 512) f16
__global__ void cvt_wck(const float* __restrict__ src, f16* __restrict__ dst) {
  int i = blockIdx.x * 256 + threadIdx.x;
  if (i >= 128 * A_) return;
  int row = i >> 9;
  dst[i] = (row < NCK_) ? (f16)src[i] : (f16)0.0f;
}

// ---------------- pack + depthwise conv ----------------
__global__ void pack_dw(const float* __restrict__ hidden, const float* __restrict__ dww,
                        f16* __restrict__ hsA, f16* __restrict__ hsC,
                        f16* __restrict__ xdw) {
  int idx = blockIdx.x * 256 + threadIdx.x;       // m*A_ + c
  if (idx >= M_ * A_) return;
  int m = idx >> 9, c = idx & (A_ - 1);
  int b = m >> 11, s = m & (S_ - 1);
  hsA[idx] = (f16)hidden[(size_t)m * HID_ + c];
  hsC[idx] = (f16)hidden[(size_t)m * HID_ + A_ + c];
  float acc = 0.0f;
#pragma unroll
  for (int j = 0; j < KW_; ++j) {
    int s2 = s + j - KW_ / 2;
    if (s2 >= 0 && s2 < S_)
      acc += hidden[((size_t)(b * S_ + s2)) * HID_ + A_ + c] * dww[c * KW_ + j];
  }
  xdw[idx] = (f16)acc;
}

// ---------------- WMMA GEMM: C[M x N] = A[M x 512] * W[N x 512]^T ----------------
// Workgroup = 8 waves = 128 M-rows x 64 N-cols. The 64x512 f16 weight tile (64KB)
// is staged into LDS once per workgroup by the Tensor Data Mover; all 8 waves
// read B-fragments from LDS (ds_load_b128), A-fragments stream from global.
// MODE 0: f16 out (B,H,S,D)   (q, k)
// MODE 1: f16 out (B,H,D,S)   (v transposed)
// MODE 2: f16 out (acc+bias)*hs_conv   (pointwise -> conv_attn)
// MODE 3: f32 out row-major   (col)
// MODE 4: f32 out logits, N<72 only    (ckl)
template<int MODE>
__global__ __launch_bounds__(256)
void gemm16x64(const f16* __restrict__ Amat, const f16* __restrict__ Wmat,
               const float* __restrict__ bias, int Ntot,
               void* __restrict__ outp, const float* __restrict__ aux) {
  __shared__ f16 ldsW[64 * A_];                    // 64 KB weight tile
  const int lane = threadIdx.x & 31;
  const int lm = lane & 15, half = lane >> 4;
  const int wave = threadIdx.x >> 5;
  const int ntiles = Ntot >> 6;
  const int mg = blockIdx.x / ntiles;              // 128-row M group
  const int nt = blockIdx.x - mg * ntiles;
  const int m0 = mg * 128 + wave * 16;
  const int n0 = nt * 64;

#if __has_builtin(__builtin_amdgcn_tensor_load_to_lds)
  if (wave == 0) {
    const unsigned long long ga =
        (unsigned long long)(uintptr_t)(Wmat + (size_t)n0 * A_);
    const unsigned la = (unsigned)(uintptr_t)(&ldsW[0]);  // LDS offset = flat[31:0]
    // D# group 0: count=1 | lds_addr | global_addr(57b) | type=2
    v4u g0 = { 1u,
               la,
               (unsigned)ga,
               ((unsigned)(ga >> 32) & 0x01FFFFFFu) | (2u << 30) };
    const unsigned dim0 = (A_ * 2) / 4;            // row length in dwords = 256
    // D# group 1: data_size=4B; tensor 256x64 dwords; tile 256x64; stride0=256
    v8i g1 = { (int)(2u << 16),                    // data_size = 2 (4 bytes)
               (int)(dim0 << 16),                  // tensor_dim0[15:0] -> bits 63:48
               (int)(64u << 16),                   // tensor_dim1 = 64 rows
               (int)(dim0 << 16),                  // tile_dim0 = 256 dwords
               (int)64,                            // tile_dim1 = 64 rows
               (int)dim0,                          // tensor_dim0_stride = 256
               0, 0 };
    v4i gz = { 0, 0, 0, 0 };
    v8i gz8 = { 0, 0, 0, 0, 0, 0, 0, 0 };
    __builtin_amdgcn_tensor_load_to_lds(g0, g1, gz, gz, gz8, 0);
    __builtin_amdgcn_s_wait_tensorcnt((short)0);
  }
  __syncthreads();
#else
  {
    const v8h* src = (const v8h*)(Wmat + (size_t)n0 * A_);
    v8h* dst = (v8h*)ldsW;
    for (int i = threadIdx.x; i < 64 * (A_ / 8); i += 256) dst[i] = src[i];
    __syncthreads();
  }
#endif

  v8f acc[4] = {};
  const f16* arow = Amat + (size_t)(m0 + lm) * A_;
  for (int kk = 0; kk < A_; kk += 32) {
    if (kk + 64 < A_) __builtin_prefetch(arow + kk + 64, 0, 0);
    // A fragment (16x32): lane row = lm; K run [kk+16h, +16)
    v16h a = ldfrag(arow + kk + half * 16, arow + kk + half * 16 + 8);
#pragma unroll
    for (int t = 0; t < 4; ++t) {
      // B fragment (32x16) from LDS: lane col = t*16+lm; K run [kk+16h, +16)
      const f16* wrow = &ldsW[(t * 16 + lm) * A_ + kk + half * 16];
      v16h b = ldfrag(wrow, wrow + 8);
      acc[t] = wmma_f16(a, b, acc[t]);
    }
  }

#pragma unroll
  for (int t = 0; t < 4; ++t) {
#pragma unroll
    for (int r = 0; r < 8; ++r) {
      const int M = m0 + r + half * 8;            // C layout: row = r + 8*half
      const int N = n0 + t * 16 + lm;             //           col = lane%16
      float v = acc[t][r];
      if (MODE == 0) {
        f16* dst = (f16*)outp;
        int b = M >> 11, s = M & (S_ - 1);
        int h = N >> 6,  d = N & 63;
        dst[(((size_t)(b * H_ + h)) * S_ + s) * D_ + d] = (f16)(v + bias[N]);
      } else if (MODE == 1) {
        f16* dst = (f16*)outp;
        int b = M >> 11, s = M & (S_ - 1);
        int h = N >> 6,  d = N & 63;
        dst[(((size_t)(b * H_ + h)) * D_ + d) * S_ + s] = (f16)(v + bias[N]);
      } else if (MODE == 2) {
        f16* dst = (f16*)outp;
        float hc = aux[(size_t)M * HID_ + A_ + N];
        dst[(size_t)M * A_ + N] = (f16)((v + bias[N]) * hc);
      } else if (MODE == 3) {
        float* dst = (float*)outp;
        dst[(size_t)M * A_ + N] = v + bias[N];
      } else {
        if (N < NCK_) {
          float* dst = (float*)outp;
          dst[(size_t)M * NCK_ + N] = v + bias[N];
        }
      }
    }
  }
}

// ---------------- softmax over K=9 per (m, h) ----------------
__global__ void softmax_ckl(const float* __restrict__ logits, float* __restrict__ probs) {
  int idx = blockIdx.x * 256 + threadIdx.x;       // (m*H_ + h); layout stride 9
  if (idx >= M_ * H_) return;
  const float* p = logits + (size_t)idx * KW_;
  float v[KW_], mx = -1e30f;
#pragma unroll
  for (int j = 0; j < KW_; ++j) { v[j] = p[j]; mx = fmaxf(mx, v[j]); }
  float sum = 0.0f;
#pragma unroll
  for (int j = 0; j < KW_; ++j) { v[j] = __expf(v[j] - mx); sum += v[j]; }
  float inv = 1.0f / sum;
  float* q = probs + (size_t)idx * KW_;
#pragma unroll
  for (int j = 0; j < KW_; ++j) q[j] = v[j] * inv;
}

// ---------------- window-gather conv output ----------------
__global__ void conv_out_k(const float* __restrict__ col, const float* __restrict__ probs,
                           float* __restrict__ out) {
  int idx = blockIdx.x * 256 + threadIdx.x;       // m*A_ + n
  if (idx >= M_ * A_) return;
  int m = idx >> 9, n = idx & (A_ - 1);
  int b = m >> 11, s = m & (S_ - 1);
  int h = n >> 6;
  const float* pr = probs + ((size_t)m * H_ + h) * KW_;
  float acc = 0.0f;
#pragma unroll
  for (int j = 0; j < KW_; ++j) {
    int s2 = s + j - KW_ / 2;
    if (s2 >= 0 && s2 < S_)
      acc += col[((size_t)(b * S_ + s2)) * A_ + n] * pr[j];
  }
  out[(size_t)m * HID_ + A_ + n] = acc;
}

// ---------------- flash attention: wave = one 16-query tile of one (b,h) ----------------
__global__ __launch_bounds__(128)
void attn_k(const f16* __restrict__ q16, const f16* __restrict__ k16,
            const f16* __restrict__ v16t, float* __restrict__ out) {
  __shared__ f16 ldsP[4][16][32];
  const int lane = threadIdx.x & 31, lm = lane & 15, half = lane >> 4;
  const int wave = threadIdx.x >> 5;
  const int wid = blockIdx.x * 4 + wave;
  const int bh = wid >> 7;                        // / (S_/16)
  const int qb = (wid & 127) << 4;
  const float scale = 0.125f;                     // 1/sqrt(64)

  // Q fragments for the two 32-wide d-chunks (row = qb+lm)
  const f16* qp = q16 + ((size_t)bh * S_ + qb + lm) * D_;
  v16h aq0 = ldfrag(qp + half * 16,      qp + half * 16 + 8);
  v16h aq1 = ldfrag(qp + 32 + half * 16, qp + 32 + half * 16 + 8);

  float mrow[8], lrow[8];
  v8f O[4] = {};
#pragma unroll
  for (int r = 0; r < 8; ++r) { mrow[r] = -1e30f; lrow[r] = 0.0f; }

  const f16* kbase = k16  + (size_t)bh * S_ * D_;
  const f16* vbase = v16t + (size_t)bh * D_ * S_;

  for (int kt = 0; kt < S_; kt += 32) {
    v8f s0 = {}, s1 = {};
    {
      const f16* kr0 = kbase + (size_t)(kt + lm) * D_;        // B col = key kt+lm
      const f16* kr1 = kbase + (size_t)(kt + 16 + lm) * D_;   // B col = key kt+16+lm
      v16h b00 = ldfrag(kr0 + half * 16,      kr0 + half * 16 + 8);
      v16h b01 = ldfrag(kr0 + 32 + half * 16, kr0 + 32 + half * 16 + 8);
      v16h b10 = ldfrag(kr1 + half * 16,      kr1 + half * 16 + 8);
      v16h b11 = ldfrag(kr1 + 32 + half * 16, kr1 + 32 + half * 16 + 8);
      s0 = wmma_f16(aq0, b00, s0);
      s0 = wmma_f16(aq1, b01, s0);
      s1 = wmma_f16(aq0, b10, s1);
      s1 = wmma_f16(aq1, b11, s1);
    }
    // online softmax; per-row stats live per-(vgpr r, lane-half) — matches C layout
    float al[8];
#pragma unroll
    for (int r = 0; r < 8; ++r) {
      float x0 = s0[r] * scale, x1 = s1[r] * scale;
      float rmx = fmaxf(x0, x1);
      rmx = fmaxf(rmx, __shfl_xor(rmx, 1, 32));
      rmx = fmaxf(rmx, __shfl_xor(rmx, 2, 32));
      rmx = fmaxf(rmx, __shfl_xor(rmx, 4, 32));
      rmx = fmaxf(rmx, __shfl_xor(rmx, 8, 32));
      float nm = fmaxf(mrow[r], rmx);
      float alpha = __expf(mrow[r] - nm);
      mrow[r] = nm;
      float p0 = __expf(x0 - nm), p1 = __expf(x1 - nm);
      float rs = p0 + p1;
      rs += __shfl_xor(rs, 1, 32);
      rs += __shfl_xor(rs, 2, 32);
      rs += __shfl_xor(rs, 4, 32);
      rs += __shfl_xor(rs, 8, 32);
      lrow[r] = lrow[r] * alpha + rs;
      al[r] = alpha;
      ldsP[wave][r + half * 8][lm]      = (f16)p0;   // reshape C-layout -> row-major
      ldsP[wave][r + half * 8][16 + lm] = (f16)p1;
    }
#pragma unroll
    for (int t = 0; t < 4; ++t)
#pragma unroll
      for (int r = 0; r < 8; ++r) O[t][r] *= al[r];
    asm volatile("s_wait_dscnt 0" ::: "memory");     // cross-lane LDS handoff
    // P as A-fragment: row = lm, key run [16h,+16)
    v16h pA = ldfrag(&ldsP[wave][lm][half * 16], &ldsP[wave][lm][half * 16 + 8]);
#pragma unroll
    for (int t = 0; t < 4; ++t) {
      // V^T fragment: B col = d = t*16+lm; key run [kt+16h, +16) contiguous
      const f16* vr = vbase + (size_t)(t * 16 + lm) * S_ + kt + half * 16;
      v16h bv = ldfrag(vr, vr + 8);
      O[t] = wmma_f16(pA, bv, O[t]);
    }
  }

  const int b = bh >> 3, h = bh & 7;
#pragma unroll
  for (int t = 0; t < 4; ++t)
#pragma unroll
    for (int r = 0; r < 8; ++r) {
      int s = qb + r + half * 8;
      out[((size_t)(b * S_ + s)) * HID_ + h * D_ + t * 16 + lm] = O[t][r] / lrow[r];
    }
}

extern "C" void kernel_launch(void* const* d_in, const int* in_sizes, int n_in,
                              void* d_out, int out_size, void* d_ws, size_t ws_size,
                              hipStream_t stream) {
  const float* hidden = (const float*)d_in[0];
  const float* Wq   = (const float*)d_in[1];
  const float* bq   = (const float*)d_in[2];
  const float* Wk   = (const float*)d_in[3];
  const float* bk   = (const float*)d_in[4];
  const float* Wv   = (const float*)d_in[5];
  const float* bv   = (const float*)d_in[6];
  const float* dww  = (const float*)d_in[7];
  const float* pw   = (const float*)d_in[8];
  const float* sepb = (const float*)d_in[9];
  const float* Wck  = (const float*)d_in[10];
  const float* bck  = (const float*)d_in[11];
  const float* Wco  = (const float*)d_in[12];
  const float* bco  = (const float*)d_in[13];
  float* out = (float*)d_out;

  char* ws = (char*)d_ws;
  size_t off = 0;
  auto alloc = [&](size_t bytes) -> void* {
    void* p = ws + off;
    off = (off + bytes + 255) & ~(size_t)255;
    return p;
  };

  f16* wq16  = (f16*)alloc((size_t)A_ * A_ * 2);
  f16* wk16  = (f16*)alloc((size_t)A_ * A_ * 2);
  f16* wv16  = (f16*)alloc((size_t)A_ * A_ * 2);
  f16* wpw16 = (f16*)alloc((size_t)A_ * A_ * 2);
  f16* wco16 = (f16*)alloc((size_t)A_ * A_ * 2);
  f16* wck16 = (f16*)alloc((size_t)128 * A_ * 2);
  f16* hsA   = (f16*)alloc((size_t)M_ * A_ * 2);
  f16* hsC   = (f16*)alloc((size_t)M_ * A_ * 2);
  f16* xdw   = (f16*)alloc((size_t)M_ * A_ * 2);
  f16* cattn = (f16*)alloc((size_t)M_ * A_ * 2);
  f16* q16   = (f16*)alloc((size_t)M_ * A_ * 2);
  f16* k16   = (f16*)alloc((size_t)M_ * A_ * 2);
  f16* v16t  = (f16*)alloc((size_t)M_ * A_ * 2);
  float* col32  = (float*)alloc((size_t)M_ * A_ * 4);
  float* cklraw = (float*)alloc((size_t)M_ * NCK_ * 4);
  float* cklS   = (float*)alloc((size_t)M_ * NCK_ * 4);

  const int cvtB = (A_ * A_ + 255) / 256;
  cvt_f16<<<cvtB, 256, 0, stream>>>(Wq,  wq16,  A_ * A_);
  cvt_f16<<<cvtB, 256, 0, stream>>>(Wk,  wk16,  A_ * A_);
  cvt_f16<<<cvtB, 256, 0, stream>>>(Wv,  wv16,  A_ * A_);
  cvt_f16<<<cvtB, 256, 0, stream>>>(pw,  wpw16, A_ * A_);
  cvt_f16<<<cvtB, 256, 0, stream>>>(Wco, wco16, A_ * A_);
  cvt_wck<<<(128 * A_ + 255) / 256, 256, 0, stream>>>(Wck, wck16);

  pack_dw<<<(M_ * A_) / 256, 256, 0, stream>>>(hidden, dww, hsA, hsC, xdw);

  const int g512 = (M_ / 128) * (A_ / 64);      // 512 blocks (8 waves each)
  gemm16x64<0><<<g512, 256, 0, stream>>>(hsA, wq16,  bq,   A_,  q16,   nullptr);
  gemm16x64<0><<<g512, 256, 0, stream>>>(hsA, wk16,  bk,   A_,  k16,   nullptr);
  gemm16x64<1><<<g512, 256, 0, stream>>>(hsA, wv16,  bv,   A_,  v16t,  nullptr);
  gemm16x64<2><<<g512, 256, 0, stream>>>(xdw, wpw16, sepb, A_,  cattn, hidden);
  gemm16x64<3><<<g512, 256, 0, stream>>>(hsC, wco16, bco,  A_,  col32, nullptr);
  const int g128 = (M_ / 128) * (128 / 64);     // 128 blocks
  gemm16x64<4><<<g128, 256, 0, stream>>>(cattn, wck16, bck, 128, cklraw, nullptr);

  softmax_ckl<<<(M_ * H_ + 255) / 256, 256, 0, stream>>>(cklraw, cklS);
  conv_out_k<<<(M_ * A_) / 256, 256, 0, stream>>>(col32, cklS, out);
  attn_k<<<(B_ * H_ * (S_ / 16)) / 4, 128, 0, stream>>>(q16, k16, v16t, out);
}
